// GraphMatchingNetwork_65738769433189
// MI455X (gfx1250) — matrix operand order repaired and checked by reading, not compiled, not run on hardware
//
#include <hip/hip_runtime.h>

#define N_NODES 100000
#define N_EDGES 1000000

typedef __attribute__((ext_vector_type(16))) __bf16 v16bf;
typedef __attribute__((ext_vector_type(8)))  float  v8f;

__device__ __forceinline__ __bf16 f2bf(float x) { return (__bf16)x; }

// ---------------------------------------------------------------------------
// Utility kernels
// ---------------------------------------------------------------------------
__global__ void k_fill(float* __restrict__ p, long n) {
    long i = (long)blockIdx.x * blockDim.x + threadIdx.x;
    long stride = (long)gridDim.x * blockDim.x;
    for (; i < n; i += stride) p[i] = 0.0f;
}

// hidden = relu(x @ w1 + b1), x: [n,4], w1: [4,64]
__global__ void k_enc1(const float* __restrict__ x, const float* __restrict__ w1,
                       const float* __restrict__ b1, float* __restrict__ h, int n) {
    __shared__ float W[256];
    __shared__ float B[64];
    for (int i = threadIdx.x; i < 256; i += blockDim.x) W[i] = w1[i];
    if (threadIdx.x < 64) B[threadIdx.x] = b1[threadIdx.x];
    __syncthreads();
    long idx = (long)blockIdx.x * blockDim.x + threadIdx.x;
    if (idx >= (long)n * 64) return;
    int i = (int)(idx >> 6), c = (int)(idx & 63);
    const float* xp = x + (size_t)i * 4;
    float v = B[c] + xp[0]*W[c] + xp[1]*W[64+c] + xp[2]*W[128+c] + xp[3]*W[192+c];
    h[idx] = fmaxf(v, 0.0f);
}

// deg count at dst (self loop added later in dinv)
__global__ void k_deg(const int* __restrict__ dst, float* __restrict__ deg, int e) {
    int i = blockIdx.x * blockDim.x + threadIdx.x;
    if (i < e) atomicAdd(&deg[dst[i]], 1.0f);
}

__global__ void k_dinv(float* __restrict__ d, int n) {
    int i = blockIdx.x * blockDim.x + threadIdx.x;
    if (i < n) d[i] = rsqrtf(d[i] + 1.0f);
}

// agg[dst] += xw[src] * dinv[src]*dinv[dst]  (one 64-thread group per edge)
__global__ void k_agg(const int* __restrict__ src, const int* __restrict__ dst,
                      const float* __restrict__ dinv, const float* __restrict__ xw,
                      float* __restrict__ agg, int e) {
    int t = blockIdx.x * 256 + threadIdx.x;
    int eidx = t >> 6;
    int f = t & 63;
    if (eidx >= e) return;
    int s = src[eidx], d = dst[eidx];
    float coef = dinv[s] * dinv[d];
    atomicAdd(&agg[(size_t)d * 64 + f], xw[(size_t)s * 64 + f] * coef);
}

// h = relu(agg + xw * dinv^2 + b)
__global__ void k_comb(const float* __restrict__ agg, const float* __restrict__ xw,
                       const float* __restrict__ dinv, const float* __restrict__ b,
                       float* __restrict__ h, int n) {
    long idx = (long)blockIdx.x * blockDim.x + threadIdx.x;
    if (idx >= (long)n * 64) return;
    int i = (int)(idx >> 6), f = (int)(idx & 63);
    float di = dinv[i];
    float v = agg[idx] + xw[idx] * di * di + b[f];
    h[idx] = fmaxf(v, 0.0f);
}

// ---------------------------------------------------------------------------
// WMMA GEMM: Y[n,64] = X[n,64] @ W[64,64] (+bias) (opt relu), bf16 WMMA f32 acc
// Block = 256 threads = 8 waves; each wave owns a 16-row stripe (4 col tiles).
// W staged in LDS already in B-fragment order: [ct][kh][lane][j] (bf16).
// Fragment maps (wave32):
//   A 16x32 bf16: lane<16: row=lane, elems j<8 -> K=kh*32+j ; j>=8 -> K=kh*32+16+(j-8)
//                 lane>=16: same rows, K offset +8
//   B 32x16 bf16: col=lane&15, elem j -> same K as A elem j (lane>=16: K offset +8)
//   C/D f32: elem r -> row = r + (lane>>4)*8, col = lane&15
// ---------------------------------------------------------------------------
__global__ void k_gemm64(const float* __restrict__ X, const float* __restrict__ W,
                         const float* __restrict__ bias, float* __restrict__ Y,
                         int n, int relu) {
    __shared__ __attribute__((aligned(32))) __bf16 Wf[4 * 2 * 32 * 16]; // 8 KB
    __shared__ float Bs[64];
    for (int idx = threadIdx.x; idx < 4096; idx += 256) {
        int k = idx >> 6, c = idx & 63;
        int kh = k >> 5, kr = k & 31, g = (kr >> 3) & 1, j = (kr & 7) + ((kr >> 4) << 3);
        int lane = g * 16 + (c & 15), ct = c >> 4;
        Wf[(((ct << 1) + kh) * 32 + lane) * 16 + j] = f2bf(W[idx]);
    }
    if (threadIdx.x < 64) Bs[threadIdx.x] = bias ? bias[threadIdx.x] : 0.0f;
    __syncthreads();

    int wave = threadIdx.x >> 5;
    int lane = threadIdx.x & 31;
    int rowBase = (blockIdx.x * 8 + wave) * 16;
    if (rowBase >= n) return;
    int g = lane >> 4, lc = lane & 15;

    const float* xr = X + (size_t)(rowBase + lc) * 64;
    v16bf a[2];
#pragma unroll
    for (int kh = 0; kh < 2; ++kh) {
        const float* p = xr + kh * 32 + g * 8;
#pragma unroll
        for (int j = 0; j < 8; ++j) {
            a[kh][j]     = f2bf(p[j]);
            a[kh][j + 8] = f2bf(p[16 + j]);
        }
    }

#pragma unroll
    for (int ct = 0; ct < 4; ++ct) {
        v8f c = {};
        v16bf b0 = *(const v16bf*)&Wf[(((ct << 1) + 0) * 32 + lane) * 16];
        v16bf b1 = *(const v16bf*)&Wf[(((ct << 1) + 1) * 32 + lane) * 16];
        c = __builtin_amdgcn_wmma_f32_16x16x32_bf16(false, a[0], false, b0, (short)0, c, false, false);
        c = __builtin_amdgcn_wmma_f32_16x16x32_bf16(false, a[1], false, b1, (short)0, c, false, false);
        int col = ct * 16 + lc;
        float bv = Bs[col];
        float* yp = Y + (size_t)rowBase * 64 + col;
#pragma unroll
        for (int r = 0; r < 8; ++r) {
            float v = c[r] + bv;
            if (relu) v = fmaxf(v, 0.0f);
            yp[(size_t)(r + g * 8) * 64] = v;
        }
    }
}

// Y[n,64] = relu([Xa | Xb] @ W[128,64] + bias)   (fusion layer, K=128)
__global__ void k_gemm128(const float* __restrict__ Xa, const float* __restrict__ Xb,
                          const float* __restrict__ W, const float* __restrict__ bias,
                          float* __restrict__ Y, int n, int relu) {
    __shared__ __attribute__((aligned(32))) __bf16 Wf[4 * 4 * 32 * 16]; // 16 KB
    __shared__ float Bs[64];
    for (int idx = threadIdx.x; idx < 8192; idx += 256) {
        int k = idx >> 6, c = idx & 63;
        int kc = k >> 5, kr = k & 31, g = (kr >> 3) & 1, j = (kr & 7) + ((kr >> 4) << 3);
        int lane = g * 16 + (c & 15), ct = c >> 4;
        Wf[(((ct << 2) + kc) * 32 + lane) * 16 + j] = f2bf(W[idx]);
    }
    if (threadIdx.x < 64) Bs[threadIdx.x] = bias[threadIdx.x];
    __syncthreads();

    int wave = threadIdx.x >> 5;
    int lane = threadIdx.x & 31;
    int rowBase = (blockIdx.x * 8 + wave) * 16;
    if (rowBase >= n) return;
    int g = lane >> 4, lc = lane & 15;

    const float* xra = Xa + (size_t)(rowBase + lc) * 64;
    const float* xrb = Xb + (size_t)(rowBase + lc) * 64;
    v16bf a[4];
#pragma unroll
    for (int kc = 0; kc < 4; ++kc) {
        const float* p = (kc < 2 ? xra : xrb) + (kc & 1) * 32 + g * 8;
#pragma unroll
        for (int j = 0; j < 8; ++j) {
            a[kc][j]     = f2bf(p[j]);
            a[kc][j + 8] = f2bf(p[16 + j]);
        }
    }

#pragma unroll
    for (int ct = 0; ct < 4; ++ct) {
        v8f c = {};
#pragma unroll
        for (int kc = 0; kc < 4; ++kc) {
            v16bf b = *(const v16bf*)&Wf[(((ct << 2) + kc) * 32 + lane) * 16];
            c = __builtin_amdgcn_wmma_f32_16x16x32_bf16(false, a[kc], false, b, (short)0, c, false, false);
        }
        int col = ct * 16 + lc;
        float bv = Bs[col];
        float* yp = Y + (size_t)rowBase * 64 + col;
#pragma unroll
        for (int r = 0; r < 8; ++r) {
            float v = c[r] + bv;
            if (relu) v = fmaxf(v, 0.0f);
            yp[(size_t)(r + g * 8) * 64] = v;
        }
    }
}

// Heads: nf[n,32] = fused @ out_w + out_b ; nt[n,2] = fused @ cls_w + cls_b
// cls_w packed as col-tile 2 (zero-padded to 16 cols) -> all ops via WMMA.
__global__ void k_heads(const float* __restrict__ X,
                        const float* __restrict__ out_w, const float* __restrict__ out_b,
                        const float* __restrict__ cls_w, const float* __restrict__ cls_b,
                        float* __restrict__ nf, float* __restrict__ nt, int n) {
    __shared__ __attribute__((aligned(32))) __bf16 Wf[3 * 2 * 32 * 16]; // 6 KB
    __shared__ float Bs[48];
    for (int idx = threadIdx.x; idx < 3072; idx += 256) Wf[idx] = f2bf(0.0f);
    if (threadIdx.x < 48) Bs[threadIdx.x] = 0.0f;
    __syncthreads();
    for (int idx = threadIdx.x; idx < 2048; idx += 256) {   // out_w 64x32
        int k = idx >> 5, c = idx & 31;
        int kh = k >> 5, kr = k & 31, g = (kr >> 3) & 1, j = (kr & 7) + ((kr >> 4) << 3);
        int lane = g * 16 + (c & 15), ct = c >> 4;
        Wf[(((ct << 1) + kh) * 32 + lane) * 16 + j] = f2bf(out_w[idx]);
    }
    for (int idx = threadIdx.x; idx < 128; idx += 256) {    // cls_w 64x2 -> ct=2
        int k = idx >> 1, c = idx & 1;
        int kh = k >> 5, kr = k & 31, g = (kr >> 3) & 1, j = (kr & 7) + ((kr >> 4) << 3);
        int lane = g * 16 + c;
        Wf[(((2 << 1) + kh) * 32 + lane) * 16 + j] = f2bf(cls_w[idx]);
    }
    if (threadIdx.x < 32) Bs[threadIdx.x] = out_b[threadIdx.x];
    if (threadIdx.x < 2)  Bs[32 + threadIdx.x] = cls_b[threadIdx.x];
    __syncthreads();

    int wave = threadIdx.x >> 5;
    int lane = threadIdx.x & 31;
    int rowBase = (blockIdx.x * 8 + wave) * 16;
    if (rowBase >= n) return;
    int g = lane >> 4, lc = lane & 15;

    const float* xr = X + (size_t)(rowBase + lc) * 64;
    v16bf a[2];
#pragma unroll
    for (int kh = 0; kh < 2; ++kh) {
        const float* p = xr + kh * 32 + g * 8;
#pragma unroll
        for (int j = 0; j < 8; ++j) {
            a[kh][j]     = f2bf(p[j]);
            a[kh][j + 8] = f2bf(p[16 + j]);
        }
    }

#pragma unroll
    for (int ct = 0; ct < 3; ++ct) {
        v8f c = {};
        v16bf b0 = *(const v16bf*)&Wf[(((ct << 1) + 0) * 32 + lane) * 16];
        v16bf b1 = *(const v16bf*)&Wf[(((ct << 1) + 1) * 32 + lane) * 16];
        c = __builtin_amdgcn_wmma_f32_16x16x32_bf16(false, a[0], false, b0, (short)0, c, false, false);
        c = __builtin_amdgcn_wmma_f32_16x16x32_bf16(false, a[1], false, b1, (short)0, c, false, false);
        if (ct < 2) {
            int col = ct * 16 + lc;
            float bv = Bs[col];
#pragma unroll
            for (int r = 0; r < 8; ++r)
                nf[(size_t)(rowBase + r + g * 8) * 32 + col] = c[r] + bv;
        } else if (lc < 2) {
            float bv = Bs[32 + lc];
#pragma unroll
            for (int r = 0; r < 8; ++r)
                nt[(size_t)(rowBase + r + g * 8) * 2 + lc] = c[r] + bv;
        }
    }
}

// ---------------------------------------------------------------------------
extern "C" void kernel_launch(void* const* d_in, const int* in_sizes, int n_in,
                              void* d_out, int out_size, void* d_ws, size_t ws_size,
                              hipStream_t stream) {
    (void)in_sizes; (void)n_in; (void)out_size; (void)ws_size;
    const float* front_x = (const float*)d_in[0];
    const float* side_x  = (const float*)d_in[1];
    const int*   f_ei    = (const int*)d_in[2];
    const int*   s_ei    = (const int*)d_in[3];
    const float* fe_w1 = (const float*)d_in[4];  const float* fe_b1 = (const float*)d_in[5];
    const float* fe_w2 = (const float*)d_in[6];  const float* fe_b2 = (const float*)d_in[7];
    const float* fe_cw = (const float*)d_in[8];  const float* fe_cb = (const float*)d_in[9];
    const float* se_w1 = (const float*)d_in[10]; const float* se_b1 = (const float*)d_in[11];
    const float* se_w2 = (const float*)d_in[12]; const float* se_b2 = (const float*)d_in[13];
    const float* se_cw = (const float*)d_in[14]; const float* se_cb = (const float*)d_in[15];
    const float* fus_w1 = (const float*)d_in[16]; const float* fus_b1 = (const float*)d_in[17];
    const float* fus_w2 = (const float*)d_in[18]; const float* fus_b2 = (const float*)d_in[19];
    const float* out_w = (const float*)d_in[20]; const float* out_b = (const float*)d_in[21];
    const float* cls_w = (const float*)d_in[22]; const float* cls_b = (const float*)d_in[23];

    const int N = N_NODES, E = N_EDGES;
    const size_t NF = (size_t)N * 64;
    float* ws = (float*)d_ws;
    float* bufF   = ws;             // [N,64] front features
    float* bufS   = ws + NF;        // [N,64] side features
    float* bufXW  = ws + 2 * NF;    // [N,64] scratch (x@W / hidden)
    float* bufAGG = ws + 3 * NF;    // [N,64] scratch (aggregate / fused)
    float* dinvF  = ws + 4 * NF;    // [N]
    float* dinvS  = dinvF + N;      // [N]
    float* nf = (float*)d_out;
    float* nt = nf + (size_t)N * 32;

    const int gemmGrid = (N / 16 + 7) / 8;           // 782 blocks, 8 waves each
    const int elemGrid = (int)((NF + 255) / 256);    // 25000
    const int edgeGrid = (E + 255) / 256;            // 3907
    const int aggGrid  = (int)(((size_t)E * 64) / 256); // 250000
    const int nGrid    = (N + 255) / 256;

    for (int br = 0; br < 2; ++br) {
        const float* x   = br ? side_x : front_x;
        const int*   ei  = br ? s_ei : f_ei;
        const float* w1  = br ? se_w1 : fe_w1;  const float* b1 = br ? se_b1 : fe_b1;
        const float* w2  = br ? se_w2 : fe_w2;  const float* b2 = br ? se_b2 : fe_b2;
        const float* cw  = br ? se_cw : fe_cw;  const float* cb = br ? se_cb : fe_cb;
        float* bufH  = br ? bufS : bufF;
        float* dinvB = br ? dinvS : dinvF;
        const int* src = ei;
        const int* dst = ei + E;

        // encoder MLP
        k_enc1<<<elemGrid, 256, 0, stream>>>(x, w1, b1, bufXW, N);
        k_gemm64<<<gemmGrid, 256, 0, stream>>>(bufXW, w2, b2, bufH, N, 0);

        // degree / normalization (once per branch)
        k_fill<<<512, 256, 0, stream>>>(dinvB, N);
        k_deg<<<edgeGrid, 256, 0, stream>>>(dst, dinvB, E);
        k_dinv<<<nGrid, 256, 0, stream>>>(dinvB, N);

        // 3 GCN layers
        for (int l = 0; l < 3; ++l) {
            k_gemm64<<<gemmGrid, 256, 0, stream>>>(bufH, cw + (size_t)l * 64 * 64,
                                                   nullptr, bufXW, N, 0);
            k_fill<<<2048, 256, 0, stream>>>(bufAGG, (long)NF);
            k_agg<<<aggGrid, 256, 0, stream>>>(src, dst, dinvB, bufXW, bufAGG, E);
            k_comb<<<elemGrid, 256, 0, stream>>>(bufAGG, bufXW, dinvB,
                                                 cb + (size_t)l * 64, bufH, N);
        }
    }

    // fusion MLP: hidden = relu([f|s] @ fus_w1 + b1) ; fused = hidden @ fus_w2 + b2
    k_gemm128<<<gemmGrid, 256, 0, stream>>>(bufF, bufS, fus_w1, fus_b1, bufXW, N, 1);
    k_gemm64<<<gemmGrid, 256, 0, stream>>>(bufXW, fus_w2, fus_b2, bufAGG, N, 0);

    // heads
    k_heads<<<gemmGrid, 256, 0, stream>>>(bufAGG, out_w, out_b, cls_w, cls_b, nf, nt, N);
}
